// QKNormCausalAttention_63256278335687
// MI455X (gfx1250) — matrix operand order repaired and checked by reading, not compiled, not run on hardware
//
#include <hip/hip_runtime.h>

#define B_    4
#define L_    2048
#define D_    1024
#define H_    16
#define HD_   64
#define MROWS (B_ * L_)

typedef __attribute__((ext_vector_type(16))) _Float16 v16h;
typedef __attribute__((ext_vector_type(8)))  float    v8f;

union Frag { uint4 u[2]; v16h h; };

// async global->LDS b128 copy (CDNA5 GLOBAL_LOAD_ASYNC_TO_LDS_B128, ASYNCcnt)
__device__ __forceinline__ void async_copy_b128(const void* g, void* lds) {
    unsigned lo = (unsigned)(uintptr_t)lds;   // low 32 bits of generic = LDS addr
    asm volatile("global_load_async_to_lds_b128 %0, %1, off"
                 :: "v"(lo), "v"(g) : "memory");
}
__device__ __forceinline__ void async_wait4() {
    asm volatile("s_wait_asynccnt 0x4" ::: "memory");
}
__device__ __forceinline__ void async_wait0() {
    asm volatile("s_wait_asynccnt 0x0" ::: "memory");
}

// ---------------------------------------------------------------------------
// fp32 -> f16 cast (vectorized by 4)
// ---------------------------------------------------------------------------
__global__ void __launch_bounds__(256)
cvt_f32_f16(const float* __restrict__ s, _Float16* __restrict__ d, int n) {
    int i = (blockIdx.x * 256 + threadIdx.x) * 4;
    if (i + 3 < n) {
        float4 v = *(const float4*)(s + i);
        d[i + 0] = (_Float16)v.x;
        d[i + 1] = (_Float16)v.y;
        d[i + 2] = (_Float16)v.z;
        d[i + 3] = (_Float16)v.w;
    }
}

// ---------------------------------------------------------------------------
// TN GEMM: Y[M,N] = A[M,K] @ W[N,K]^T   (A,W f16 row-major, f32 accumulate)
// block: 256 thr = 8 waves; block tile 128(M) x 128(N); wave tile 32 x 64
// double-buffered tiles staged via GLOBAL_LOAD_ASYNC_TO_LDS_B128 so the DMA
// for tile k+1 overlaps the WMMA stream on tile k.
// ---------------------------------------------------------------------------
template <bool OUT_F32>
__global__ void __launch_bounds__(256)
gemm_tn(const _Float16* __restrict__ A, const _Float16* __restrict__ W,
        void* __restrict__ Yv, int Kdim, int Ndim) {
    __shared__ _Float16 lA[2][128 * 32];
    __shared__ _Float16 lW[2][128 * 32];

    const int tid  = threadIdx.x;
    const int lane = tid & 31;
    const int wave = tid >> 5;
    const int half = lane >> 4;
    const int l16  = lane & 15;
    const int wm   = wave >> 1;          // 0..3  (M sub-tile)
    const int wn   = wave & 1;           // 0..1  (N sub-tile)
    const int mB   = blockIdx.y * 128;
    const int nB   = blockIdx.x * 128;

    const int rowL = tid >> 2;           // 0..63  (+64 for second chunk)
    const int c4   = tid & 3;

    auto issue = [&](int k0, int buf) {
        async_copy_b128(A + (size_t)(mB + rowL) * Kdim + k0 + c4 * 8,
                        lA[buf] + rowL * 32 + c4 * 8);
        async_copy_b128(A + (size_t)(mB + rowL + 64) * Kdim + k0 + c4 * 8,
                        lA[buf] + (rowL + 64) * 32 + c4 * 8);
        async_copy_b128(W + (size_t)(nB + rowL) * Kdim + k0 + c4 * 8,
                        lW[buf] + rowL * 32 + c4 * 8);
        async_copy_b128(W + (size_t)(nB + rowL + 64) * Kdim + k0 + c4 * 8,
                        lW[buf] + (rowL + 64) * 32 + c4 * 8);
    };

    v8f acc[2][4] = {};

    issue(0, 0);                          // prologue: tile 0 -> buffer 0
    int it = 0;
    for (int k0 = 0; k0 < Kdim; k0 += 32, ++it) {
        const int cur = it & 1;
        __syncthreads();                  // prior reads of buf cur^1 retired
        if (k0 + 32 < Kdim) {
            issue(k0 + 32, cur ^ 1);      // DMA next tile while computing
            async_wait4();                // oldest 4 (= buf cur) complete
        } else {
            async_wait0();
        }
        __syncthreads();                  // buf cur visible block-wide

        const _Float16* la = lA[cur];
        const _Float16* lw = lW[cur];
        Frag a[2];
#pragma unroll
        for (int s = 0; s < 2; ++s) {
            const uint4* arow = (const uint4*)(la + (wm * 32 + s * 16 + l16) * 32);
            a[s].u[0] = arow[half];
            a[s].u[1] = arow[half + 2];
        }
#pragma unroll
        for (int f = 0; f < 4; ++f) {
            Frag b;
            const uint4* brow = (const uint4*)(lw + (wn * 64 + f * 16 + l16) * 32);
            b.u[0] = brow[half];
            b.u[1] = brow[half + 2];
#pragma unroll
            for (int s = 0; s < 2; ++s)
                acc[s][f] = __builtin_amdgcn_wmma_f32_16x16x32_f16(
                    false, a[s].h, false, b.h, (short)0, acc[s][f], false, false);
        }
    }

#pragma unroll
    for (int s = 0; s < 2; ++s)
#pragma unroll
        for (int f = 0; f < 4; ++f)
#pragma unroll
            for (int i = 0; i < 8; ++i) {
                size_t row = (size_t)(mB + wm * 32 + s * 16 + i + half * 8);
                size_t col = (size_t)(nB + wn * 64 + f * 16 + l16);
                if (OUT_F32)
                    ((float*)Yv)[row * Ndim + col] = acc[s][f][i];
                else
                    ((_Float16*)Yv)[row * Ndim + col] = (_Float16)acc[s][f][i];
            }
}

// ---------------------------------------------------------------------------
// Per-head LayerNorm over HD=64, in place on f16 buffer [MROWS, D]
// ---------------------------------------------------------------------------
__global__ void __launch_bounds__(256)
ln_heads(_Float16* __restrict__ q, const float* __restrict__ gamma,
         const float* __restrict__ beta) {
    int wave = threadIdx.x >> 5, lane = threadIdx.x & 31;
    int rh  = blockIdx.x * 8 + wave;
    int row = rh / H_, h = rh % H_;
    _Float16* p = q + (size_t)row * D_ + h * HD_ + lane * 2;
    float v0 = (float)p[0], v1 = (float)p[1];
    float s  = v0 + v1;
    float sq = v0 * v0 + v1 * v1;
#pragma unroll
    for (int o = 16; o >= 1; o >>= 1) {
        s  += __shfl_xor(s, o, 32);
        sq += __shfl_xor(sq, o, 32);
    }
    float mu  = s * (1.0f / 64.0f);
    float var = sq * (1.0f / 64.0f) - mu * mu;
    float r   = rsqrtf(var + 1e-5f);
    float g0 = gamma[lane * 2], g1 = gamma[lane * 2 + 1];
    float b0 = beta[lane * 2],  b1 = beta[lane * 2 + 1];
    p[0] = (_Float16)((v0 - mu) * r * g0 + b0);
    p[1] = (_Float16)((v1 - mu) * r * g1 + b1);
}

// ---------------------------------------------------------------------------
// Causal flash attention. grid (L/128, B*H); 8 waves/block; each wave owns a
// 16-row q tile, iterates 32-key tiles with streaming softmax.
// ---------------------------------------------------------------------------
__global__ void __launch_bounds__(256)
attn_causal(const _Float16* __restrict__ Q, const _Float16* __restrict__ K,
            const _Float16* __restrict__ V, _Float16* __restrict__ ctx) {
    __shared__ _Float16 lVt[8][HD_ * 32];  // per-wave V^T (d-major)
    __shared__ _Float16 lP[8][16 * 32];    // per-wave P staging

    const int lane = threadIdx.x & 31;
    const int wave = threadIdx.x >> 5;
    const int half = lane >> 4;
    const int l16  = lane & 15;
    const int bh = blockIdx.y;
    const int b  = bh / H_, h = bh % H_;
    const int q0 = blockIdx.x * 128 + wave * 16;
    const size_t rowBase = (size_t)b * L_;

    // Q fragments (16x64 A-layout, two K=32 fragments); fold in 1/sqrt(HD)
    Frag qa[2];
    {
        const uint4* qrow = (const uint4*)(Q + (rowBase + q0 + l16) * D_ + h * HD_);
        qa[0].u[0] = qrow[half];
        qa[0].u[1] = qrow[half + 2];
        qa[1].u[0] = qrow[half + 4];
        qa[1].u[1] = qrow[half + 6];
        qa[0].h = qa[0].h * (_Float16)0.125f;   // exact: power of two
        qa[1].h = qa[1].h * (_Float16)0.125f;
    }

    float mrow[8], lrow[8];
    v8f cacc[4] = {};
#pragma unroll
    for (int i = 0; i < 8; ++i) { mrow[i] = -1e30f; lrow[i] = 0.0f; }

    _Float16* vt = lVt[wave];
    _Float16* lp = lP[wave];
    const int ntiles = (q0 + 16 + 31) >> 5;

    for (int t = 0; t < ntiles; ++t) {
        const int kt = t * 32;

        // prefetch next tile's K/V rows (global_prefetch_b8)
        if (t + 1 < ntiles) {
            __builtin_prefetch(K + (rowBase + kt + 32 + lane) * D_ + h * HD_, 0, 0);
            __builtin_prefetch(V + (rowBase + kt + 32 + lane) * D_ + h * HD_, 0, 0);
        }

        // stage V^T: each lane transposes one 8(key) x 8(d) patch.
        // v_perm_b32 merges f16 halves from two rows -> ds_store_b128 rows.
        {
            const int kg = (lane >> 3) * 8;     // key group base (0,8,16,24)
            const int d8 = (lane & 7) * 8;      // d chunk
            union { uint4 u; unsigned w[4]; } r[8];
#pragma unroll
            for (int j = 0; j < 8; ++j)
                r[j].u = *(const uint4*)(V + (rowBase + kt + kg + j) * D_ + h * HD_ + d8);
#pragma unroll
            for (int e = 0; e < 8; ++e) {
                const int wi = e >> 1;
                const unsigned sel = (e & 1) ? 0x07060302u : 0x05040100u;
                uint4 o;
                o.x = __builtin_amdgcn_perm(r[1].w[wi], r[0].w[wi], sel);
                o.y = __builtin_amdgcn_perm(r[3].w[wi], r[2].w[wi], sel);
                o.z = __builtin_amdgcn_perm(r[5].w[wi], r[4].w[wi], sel);
                o.w = __builtin_amdgcn_perm(r[7].w[wi], r[6].w[wi], sel);
                *(uint4*)(vt + (d8 + e) * 32 + kg) = o;
            }
        }

        // S = Q K^T  (two 16-key groups, K=64 contraction = 2 chained WMMA)
        v8f s[2] = {};
#pragma unroll
        for (int g = 0; g < 2; ++g) {
            Frag kb0, kb1;
            const uint4* krow =
                (const uint4*)(K + (rowBase + kt + g * 16 + l16) * D_ + h * HD_);
            kb0.u[0] = krow[half];
            kb0.u[1] = krow[half + 2];
            kb1.u[0] = krow[half + 4];
            kb1.u[1] = krow[half + 6];
            s[g] = __builtin_amdgcn_wmma_f32_16x16x32_f16(
                false, qa[0].h, false, kb0.h, (short)0, s[g], false, false);
            s[g] = __builtin_amdgcn_wmma_f32_16x16x32_f16(
                false, qa[1].h, false, kb1.h, (short)0, s[g], false, false);
        }

        // causal mask (C layout: row = i + half*8, col = g*16 + l16)
#pragma unroll
        for (int g = 0; g < 2; ++g) {
            int key = kt + g * 16 + l16;
#pragma unroll
            for (int i = 0; i < 8; ++i) {
                int qr = q0 + i + half * 8;
                s[g][i] = (key <= qr) ? s[g][i] : -1e30f;
            }
        }

        // streaming softmax
        float mnew[8];
#pragma unroll
        for (int i = 0; i < 8; ++i) {
            float t2 = fmaxf(s[0][i], s[1][i]);
#pragma unroll
            for (int o = 8; o >= 1; o >>= 1)
                t2 = fmaxf(t2, __shfl_xor(t2, o, 32));
            mnew[i] = fmaxf(mrow[i], t2);
        }
#pragma unroll
        for (int i = 0; i < 8; ++i) {
            float corr = __expf(mrow[i] - mnew[i]);
            mrow[i] = mnew[i];
            lrow[i] *= corr;
#pragma unroll
            for (int f = 0; f < 4; ++f) cacc[f][i] *= corr;
        }
#pragma unroll
        for (int i = 0; i < 8; ++i) {
            float p0 = __expf(s[0][i] - mrow[i]);
            float p1 = __expf(s[1][i] - mrow[i]);
            s[0][i] = p0; s[1][i] = p1;
            float rs = p0 + p1;
#pragma unroll
            for (int o = 8; o >= 1; o >>= 1) rs += __shfl_xor(rs, o, 32);
            lrow[i] += rs;
        }

        // P -> LDS (row-major 16x32), reload as A fragment
#pragma unroll
        for (int g = 0; g < 2; ++g)
#pragma unroll
            for (int i = 0; i < 8; ++i)
                lp[(i + half * 8) * 32 + g * 16 + l16] = (_Float16)s[g][i];

        Frag pa;
        const uint4* prow = (const uint4*)(lp + l16 * 32);
        pa.u[0] = prow[half];
        pa.u[1] = prow[half + 2];

        // ctx += P @ V   (B fragments from V^T rows, contiguous ds_read_b128)
#pragma unroll
        for (int f = 0; f < 4; ++f) {
            Frag vb;
            const uint4* vrow = (const uint4*)(vt + (f * 16 + l16) * 32);
            vb.u[0] = vrow[half];
            vb.u[1] = vrow[half + 2];
            cacc[f] = __builtin_amdgcn_wmma_f32_16x16x32_f16(
                false, pa.h, false, vb.h, (short)0, cacc[f], false, false);
        }
    }

    // finalize: divide by row sums, store f16 ctx
#pragma unroll
    for (int f = 0; f < 4; ++f)
#pragma unroll
        for (int i = 0; i < 8; ++i) {
            float v = cacc[f][i] / lrow[i];
            ctx[(rowBase + q0 + i + half * 8) * D_ + h * HD_ + f * 16 + l16] =
                (_Float16)v;
        }
}

// ---------------------------------------------------------------------------
extern "C" void kernel_launch(void* const* d_in, const int* in_sizes, int n_in,
                              void* d_out, int out_size, void* d_ws, size_t ws_size,
                              hipStream_t stream) {
    const float* x  = (const float*)d_in[0];
    // d_in[1] = causal mask, synthesized on the fly instead
    const float* Wq = (const float*)d_in[2];
    const float* Wk = (const float*)d_in[3];
    const float* Wv = (const float*)d_in[4];
    const float* Wo = (const float*)d_in[5];
    const float* qg = (const float*)d_in[6];
    const float* qb = (const float*)d_in[7];
    const float* kg = (const float*)d_in[8];
    const float* kb = (const float*)d_in[9];
    float* out = (float*)d_out;

    char* ws = (char*)d_ws;
    _Float16* xh  = (_Float16*)ws; ws += (size_t)MROWS * D_ * 2;
    _Float16* wqh = (_Float16*)ws; ws += (size_t)D_ * D_ * 2;
    _Float16* wkh = (_Float16*)ws; ws += (size_t)D_ * D_ * 2;
    _Float16* wvh = (_Float16*)ws; ws += (size_t)D_ * D_ * 2;
    _Float16* woh = (_Float16*)ws; ws += (size_t)D_ * D_ * 2;
    _Float16* Qh  = (_Float16*)ws; ws += (size_t)MROWS * D_ * 2;
    _Float16* Kh  = (_Float16*)ws; ws += (size_t)MROWS * D_ * 2;
    _Float16* Vh  = (_Float16*)ws; ws += (size_t)MROWS * D_ * 2;
    _Float16* Ch  = (_Float16*)ws; ws += (size_t)MROWS * D_ * 2;

    const int nx = MROWS * D_;
    const int nw = D_ * D_;
    cvt_f32_f16<<<nx / 1024, 256, 0, stream>>>(x,  xh,  nx);
    cvt_f32_f16<<<nw / 1024, 256, 0, stream>>>(Wq, wqh, nw);
    cvt_f32_f16<<<nw / 1024, 256, 0, stream>>>(Wk, wkh, nw);
    cvt_f32_f16<<<nw / 1024, 256, 0, stream>>>(Wv, wvh, nw);
    cvt_f32_f16<<<nw / 1024, 256, 0, stream>>>(Wo, woh, nw);

    dim3 ggrid(D_ / 128, MROWS / 128);
    gemm_tn<false><<<ggrid, 256, 0, stream>>>(xh, wqh, (void*)Qh, D_, D_);
    gemm_tn<false><<<ggrid, 256, 0, stream>>>(xh, wkh, (void*)Kh, D_, D_);
    gemm_tn<false><<<ggrid, 256, 0, stream>>>(xh, wvh, (void*)Vh, D_, D_);

    ln_heads<<<MROWS * H_ / 8, 256, 0, stream>>>(Qh, qg, qb);
    ln_heads<<<MROWS * H_ / 8, 256, 0, stream>>>(Kh, kg, kb);

    attn_causal<<<dim3(L_ / 128, B_ * H_), 256, 0, stream>>>(Qh, Kh, Vh, Ch);

    gemm_tn<true><<<ggrid, 256, 0, stream>>>(Ch, woh, (void*)out, D_, D_);
}
